// GNN_graphpred_67327907332507
// MI455X (gfx1250) — compile-verified, matrix operand order checked by reading, not run on hardware
//
#include <hip/hip_runtime.h>

typedef __attribute__((ext_vector_type(16))) __bf16 v16bf;
typedef __attribute__((ext_vector_type(8)))  __bf16 bf16x8;
typedef __attribute__((ext_vector_type(4)))  __bf16 bf16x4;
typedef __attribute__((ext_vector_type(8)))  float  v8f;
typedef __attribute__((ext_vector_type(4)))  float  f32x4;

#define NN 50000
#define MP 50048          // rows padded to multiple of 128 (391 * 128)
#define EE 100000
#define GG 2000
#define DD 300
#define DP 320            // D padded to multiple of 64
#define D2 600
#define D2P 640           // 2D padded to multiple of 64
#define LL 5
#define TT 10

// ---------------------------------------------------------------------------
// Weight prep: transpose + pad + convert to bf16.  WT[n][k] = W[k][n].
// ---------------------------------------------------------------------------
__global__ void k_prep_w(const float* __restrict__ W, __bf16* __restrict__ WT,
                         int K, int Nc, int Kpad, int Npad) {
    int idx = blockIdx.x * blockDim.x + threadIdx.x;
    if (idx >= Npad * Kpad) return;
    int n = idx / Kpad, k = idx - n * Kpad;
    float v = (n < Nc && k < K) ? W[(long long)k * Nc + n] : 0.0f;
    WT[idx] = (__bf16)v;
}

__global__ void k_prep_b(const float* __restrict__ b, float* __restrict__ bp,
                         int Nc, int Npad) {
    int idx = blockIdx.x * blockDim.x + threadIdx.x;
    if (idx >= Npad) return;
    bp[idx] = (idx < Nc) ? b[idx] : 0.0f;
}

// ---------------------------------------------------------------------------
// Node embedding gather into padded [MP x DP] buffer (pad rows/cols = 0).
// ---------------------------------------------------------------------------
__global__ void k_embed(const int* __restrict__ x, const float* __restrict__ xemb1,
                        const float* __restrict__ xemb2, float* __restrict__ h) {
    int idx = blockIdx.x * blockDim.x + threadIdx.x;     // over MP * (DP/4)
    if (idx >= MP * (DP / 4)) return;
    int i = idx / (DP / 4);
    int c = (idx - i * (DP / 4)) * 4;
    f32x4 v = {0.0f, 0.0f, 0.0f, 0.0f};
    if (i < NN && c < DD) {
        int a0 = x[2 * i], a1 = x[2 * i + 1];
        f32x4 e1 = *(const f32x4*)(xemb1 + (long long)a0 * DD + c);
        f32x4 e2 = *(const f32x4*)(xemb2 + (long long)a1 * DD + c);
        v = e1 + e2;
    }
    *(f32x4*)(h + (long long)i * DP + c) = v;
}

// ---------------------------------------------------------------------------
// agg = h + self-loop edge embedding (eemb1_l[4] + eemb2_l[0]); pads -> 0.
// ---------------------------------------------------------------------------
__global__ void k_agg_init(const float* __restrict__ h, const float* __restrict__ e1l,
                           const float* __restrict__ e2l, float* __restrict__ agg) {
    int idx = blockIdx.x * blockDim.x + threadIdx.x;
    if (idx >= MP * (DP / 4)) return;
    int i = idx / (DP / 4);
    int c = (idx - i * (DP / 4)) * 4;
    f32x4 v = {0.0f, 0.0f, 0.0f, 0.0f};
    if (i < NN && c < DD) {
        f32x4 hv = *(const f32x4*)(h + (long long)i * DP + c);
        f32x4 s1 = *(const f32x4*)(e1l + 4 * DD + c);
        f32x4 s2 = *(const f32x4*)(e2l + c);
        v = hv + s1 + s2;
    }
    *(f32x4*)(agg + (long long)i * DP + c) = v;
}

// ---------------------------------------------------------------------------
// Edge scatter-add: agg[dst] += h[src] + eemb1_l[ea0] + eemb2_l[ea1]
// 75 threads/edge, 4 channels each (vector gathers + global_atomic_add_f32).
// ---------------------------------------------------------------------------
__global__ void k_scatter(const int* __restrict__ ei, const int* __restrict__ ea,
                          const float* __restrict__ h, const float* __restrict__ e1l,
                          const float* __restrict__ e2l, float* __restrict__ agg) {
    const int CH = DD / 4;                                // 75
    int idx = blockIdx.x * blockDim.x + threadIdx.x;
    if (idx >= EE * CH) return;
    int e = idx / CH;
    int c = (idx - e * CH) * 4;
    int s  = ei[e];
    int t  = ei[EE + e];
    int a0 = ea[2 * e];
    int a1 = ea[2 * e + 1];
    f32x4 hv = *(const f32x4*)(h   + (long long)s  * DP + c);
    f32x4 p1 = *(const f32x4*)(e1l + (long long)a0 * DD + c);
    f32x4 p2 = *(const f32x4*)(e2l + (long long)a1 * DD + c);
    f32x4 m  = hv + p1 + p2;
    float* po = agg + (long long)t * DP + c;
    atomicAdd(po + 0, m[0]);
    atomicAdd(po + 1, m[1]);
    atomicAdd(po + 2, m[2]);
    atomicAdd(po + 3, m[3]);
}

// ---------------------------------------------------------------------------
// Branch-free WMMA GEMM on padded buffers:
//   C[Mp x Npad] = act(A[Mp x Kpad] @ WT^T + bias)
// A: f32, stride Kpad (pad cols zero).  WT: bf16 [Npad x Kpad] (transposed,
// zero-padded).  Block = 256 threads = 8 waves, tile 128x64; each wave owns a
// 16x64 slab = 4 x v_wmma_f32_16x16x32_bf16 per K-step with A-fragment reuse.
// LDS rows padded to 40 bf16 (80B, 16B aligned) to stagger banks.
// ---------------------------------------------------------------------------
#define BM 128
#define BN 64
#define BK 32
#define LDA 40

__global__ __launch_bounds__(256)
void k_gemm(const float* __restrict__ A, const __bf16* __restrict__ WT,
            const float* __restrict__ bias, float* __restrict__ C,
            int Kpad, int Npad, int do_relu) {
    __shared__ __align__(16) __bf16 As [BM][LDA];
    __shared__ __align__(16) __bf16 Bts[BN][LDA];         // n-major, k contiguous

    const int tid  = threadIdx.x;
    const int lane = tid & 31;
    const int wv   = tid >> 5;
    const int tm   = wv * 16;                             // 0..112
    const int m0   = blockIdx.x * BM;
    const int n0   = blockIdx.y * BN;

    v8f acc0 = {};
    v8f acc1 = {};
    v8f acc2 = {};
    v8f acc3 = {};

    const int ksteps = Kpad >> 5;
    for (int ks = 0; ks < ksteps; ++ks) {
        const int k0 = ks * BK;

        // A tile: 128x32 f32 -> bf16.  1024 float4 chunks, 4 per thread.
#pragma unroll
        for (int j = 0; j < 4; ++j) {
            int q    = tid + 256 * j;
            int row  = q >> 3;
            int col4 = (q & 7) * 4;
            f32x4 v = *(const f32x4*)(A + (long long)(m0 + row) * Kpad + k0 + col4);
            bf16x4 o;
            o[0] = (__bf16)v[0]; o[1] = (__bf16)v[1];
            o[2] = (__bf16)v[2]; o[3] = (__bf16)v[3];
            *(bf16x4*)&As[row][col4] = o;
        }
        // B tile: 64 n-rows x 32 k, bf16 b128 loads, 1 per thread.
        {
            int row = tid >> 2;
            int c8  = (tid & 3) * 8;
            bf16x8 v = *(const bf16x8*)(WT + (long long)(n0 + row) * Kpad + k0 + c8);
            *(bf16x8*)&Bts[row][c8] = v;
        }
        __syncthreads();

        // Fragments per CDNA5 16-bit layout (ISA 7.12.2): lane<16 holds
        // K {0..7,16..23}, lane>=16 holds K {8..15,24..31} -> two b128 loads.
        const int row = lane & 15;
        const int kh  = (lane >> 4) * 8;
        bf16x8 alo = *(const bf16x8*)&As[tm + row][kh];
        bf16x8 ahi = *(const bf16x8*)&As[tm + row][kh + 16];
        v16bf af = __builtin_shufflevector(alo, ahi,
                    0,1,2,3,4,5,6,7,8,9,10,11,12,13,14,15);

        bf16x8 b0l = *(const bf16x8*)&Bts[row][kh];
        bf16x8 b0h = *(const bf16x8*)&Bts[row][kh + 16];
        v16bf bf0 = __builtin_shufflevector(b0l, b0h,
                    0,1,2,3,4,5,6,7,8,9,10,11,12,13,14,15);
        bf16x8 b1l = *(const bf16x8*)&Bts[16 + row][kh];
        bf16x8 b1h = *(const bf16x8*)&Bts[16 + row][kh + 16];
        v16bf bf1 = __builtin_shufflevector(b1l, b1h,
                    0,1,2,3,4,5,6,7,8,9,10,11,12,13,14,15);
        bf16x8 b2l = *(const bf16x8*)&Bts[32 + row][kh];
        bf16x8 b2h = *(const bf16x8*)&Bts[32 + row][kh + 16];
        v16bf bf2 = __builtin_shufflevector(b2l, b2h,
                    0,1,2,3,4,5,6,7,8,9,10,11,12,13,14,15);
        bf16x8 b3l = *(const bf16x8*)&Bts[48 + row][kh];
        bf16x8 b3h = *(const bf16x8*)&Bts[48 + row][kh + 16];
        v16bf bf3 = __builtin_shufflevector(b3l, b3h,
                    0,1,2,3,4,5,6,7,8,9,10,11,12,13,14,15);

        acc0 = __builtin_amdgcn_wmma_f32_16x16x32_bf16(
            false, af, false, bf0, (short)0, acc0, false, false);
        acc1 = __builtin_amdgcn_wmma_f32_16x16x32_bf16(
            false, af, false, bf1, (short)0, acc1, false, false);
        acc2 = __builtin_amdgcn_wmma_f32_16x16x32_bf16(
            false, af, false, bf2, (short)0, acc2, false, false);
        acc3 = __builtin_amdgcn_wmma_f32_16x16x32_bf16(
            false, af, false, bf3, (short)0, acc3, false, false);
        __syncthreads();
    }

    // Epilogue: unconditional (zero-padded bias/weights make pad cols 0).
    const int col = lane & 15;
    const int mhi = (lane >> 4) << 3;
    float bi0 = bias[n0 + col];
    float bi1 = bias[n0 + 16 + col];
    float bi2 = bias[n0 + 32 + col];
    float bi3 = bias[n0 + 48 + col];
#pragma unroll
    for (int r = 0; r < 8; ++r) {
        int grow = m0 + tm + r + mhi;
        float v0 = acc0[r] + bi0;
        float v1 = acc1[r] + bi1;
        float v2 = acc2[r] + bi2;
        float v3 = acc3[r] + bi3;
        if (do_relu) {
            v0 = fmaxf(v0, 0.0f); v1 = fmaxf(v1, 0.0f);
            v2 = fmaxf(v2, 0.0f); v3 = fmaxf(v3, 0.0f);
        }
        long long base = (long long)grow * Npad + n0;
        C[base + col]      = v0;
        C[base + 16 + col] = v1;
        C[base + 32 + col] = v2;
        C[base + 48 + col] = v3;
    }
}

// ---------------------------------------------------------------------------
// BatchNorm stats, coalesced: blockDim = DP (320) threads <-> columns.
// Each block reduces 64 rows, then f32-atomics into gacc[0..DP)=sum,
// gacc[512..512+DP)=sumsq.
// ---------------------------------------------------------------------------
__global__ __launch_bounds__(DP)
void k_bn_partial(const float* __restrict__ h2, float* __restrict__ gacc) {
    const int t    = threadIdx.x;
    const int base = blockIdx.x * 64;
    float s = 0.0f, q = 0.0f;
    for (int r = 0; r < 64; ++r) {
        int row = base + r;
        if (row < NN) {
            float v = h2[(long long)row * DP + t];
            s += v;
            q += v * v;
        }
    }
    atomicAdd(&gacc[t], s);
    atomicAdd(&gacc[512 + t], q);
}

__global__ void k_bn_final(const float* __restrict__ gacc, float* __restrict__ mu,
                           float* __restrict__ rstd) {
    int d = threadIdx.x;
    if (d >= DD) return;
    float m   = gacc[d] / (float)NN;
    float var = gacc[512 + d] / (float)NN - m * m;
    mu[d]   = m;
    rstd[d] = rsqrtf(var + 1e-5f);
}

// ---------------------------------------------------------------------------
// BatchNorm apply (+ optional ReLU). Reads padded [MP x DP], writes dense
// rows < NN with given output stride (DP for ws h, DD for d_out node_rep).
// ---------------------------------------------------------------------------
__global__ void k_bn_apply(const float* __restrict__ h2, const float* __restrict__ mu,
                           const float* __restrict__ rstd, const float* __restrict__ gam,
                           const float* __restrict__ bet, float* __restrict__ out,
                           int ostride, int do_relu) {
    const int CH = DD / 4;
    int idx = blockIdx.x * blockDim.x + threadIdx.x;
    if (idx >= NN * CH) return;
    int i = idx / CH;
    int c = (idx - i * CH) * 4;
    f32x4 v = *(const f32x4*)(h2 + (long long)i * DP + c);
    f32x4 mv = *(const f32x4*)(mu + c);
    f32x4 rv = *(const f32x4*)(rstd + c);
    f32x4 gv = *(const f32x4*)(gam + c);
    f32x4 bv = *(const f32x4*)(bet + c);
    f32x4 o = gv * (v - mv) * rv + bv;
    if (do_relu) {
        o[0] = fmaxf(o[0], 0.0f); o[1] = fmaxf(o[1], 0.0f);
        o[2] = fmaxf(o[2], 0.0f); o[3] = fmaxf(o[3], 0.0f);
    }
    *(f32x4*)(out + (long long)i * ostride + c) = o;
}

// ---------------------------------------------------------------------------
// Pooling + prediction head
// ---------------------------------------------------------------------------
__global__ void k_zero(float* __restrict__ p, int n) {
    int idx = blockIdx.x * blockDim.x + threadIdx.x;
    if (idx < n) p[idx] = 0.0f;
}

__global__ void k_count(const int* __restrict__ batch, float* __restrict__ cnt) {
    int i = blockIdx.x * blockDim.x + threadIdx.x;
    if (i < NN) atomicAdd(&cnt[batch[i]], 1.0f);
}

__global__ void k_pool_sum(const int* __restrict__ batch, const float* __restrict__ node,
                           float* __restrict__ gsum) {
    const int CH = DD / 4;
    int idx = blockIdx.x * blockDim.x + threadIdx.x;
    if (idx >= NN * CH) return;
    int i = idx / CH;
    int c = (idx - i * CH) * 4;
    int g = batch[i];
    f32x4 v = *(const f32x4*)(node + (long long)i * DD + c);
    float* pg = gsum + (long long)g * DD + c;
    atomicAdd(pg + 0, v[0]);
    atomicAdd(pg + 1, v[1]);
    atomicAdd(pg + 2, v[2]);
    atomicAdd(pg + 3, v[3]);
}

__global__ void k_pool_div(const float* __restrict__ gsum, const float* __restrict__ cnt,
                           float* __restrict__ grep) {
    int idx = blockIdx.x * blockDim.x + threadIdx.x;
    if (idx >= GG * DD) return;
    int g = idx / DD;
    grep[idx] = gsum[idx] / fmaxf(cnt[g], 1.0f);
}

__global__ void k_pred(const float* __restrict__ grep, const float* __restrict__ predW,
                       const float* __restrict__ predb, float* __restrict__ pred) {
    int idx = blockIdx.x * blockDim.x + threadIdx.x;
    if (idx >= GG * TT) return;
    int g = idx / TT, t = idx - g * TT;
    float s = predb[t];
    const float* row = grep + (long long)g * DD;
    for (int d = 0; d < DD; ++d) s += row[d] * predW[d * TT + t];
    pred[idx] = s;
}

// ---------------------------------------------------------------------------
// Host launcher
// ---------------------------------------------------------------------------
static inline unsigned cdiv(long long a, long long b) { return (unsigned)((a + b - 1) / b); }

extern "C" void kernel_launch(void* const* d_in, const int* in_sizes, int n_in,
                              void* d_out, int out_size, void* d_ws, size_t ws_size,
                              hipStream_t stream) {
    (void)in_sizes; (void)n_in; (void)out_size; (void)ws_size;

    const float* xemb1 = (const float*)d_in[8];
    const float* xemb2 = (const float*)d_in[9];
    const float* eemb1 = (const float*)d_in[10];  // [L,6,D]
    const float* eemb2 = (const float*)d_in[11];  // [L,3,D]
    const float* W1    = (const float*)d_in[12];  // [L,D,2D]
    const float* b1    = (const float*)d_in[13];  // [L,2D]
    const float* W2    = (const float*)d_in[14];  // [L,2D,D]
    const float* b2    = (const float*)d_in[15];  // [L,D]
    const float* gam   = (const float*)d_in[16];  // [L,D]
    const float* bet   = (const float*)d_in[17];  // [L,D]
    const float* predW = (const float*)d_in[18];  // [D,T]
    const float* predb = (const float*)d_in[19];  // [T]

    // workspace carve-out (floats unless noted)
    float* ws    = (float*)d_ws;
    float* h     = ws;                                   // MP*DP
    float* agg   = h    + (size_t)MP * DP;               // MP*DP (gemm2 output)
    float* h2a   = agg  + (size_t)MP * DP;               // MP*D2P
    float* mu    = h2a  + (size_t)MP * D2P;              // 512
    float* rstd  = mu + 512;                             // 512
    float* gacc  = rstd + 512;                           // 1024 (sum | sumsq)
    float* gsum  = gacc + 1024;                          // GG*DD
    float* gcnt  = gsum + (size_t)GG * DD;               // GG
    float* bp1   = gcnt + GG;                            // LL*D2P
    float* bp2   = bp1 + (size_t)LL * D2P;               // LL*DP
    __bf16* w1t  = (__bf16*)(bp2 + (size_t)LL * DP);     // LL*D2P*DP bf16
    __bf16* w2t  = w1t + (size_t)LL * D2P * DP;          // LL*DP*D2P bf16

    const int TB = 256;

    // One-time per-call weight prep (transpose + pad + bf16)
    for (int l = 0; l < LL; ++l) {
        k_prep_w<<<cdiv((long long)D2P * DP, TB), TB, 0, stream>>>(
            W1 + (size_t)l * DD * D2, w1t + (size_t)l * D2P * DP, DD, D2, DP, D2P);
        k_prep_w<<<cdiv((long long)DP * D2P, TB), TB, 0, stream>>>(
            W2 + (size_t)l * D2 * DD, w2t + (size_t)l * DP * D2P, D2, DD, D2P, DP);
        k_prep_b<<<cdiv(D2P, TB), TB, 0, stream>>>(b1 + (size_t)l * D2, bp1 + (size_t)l * D2P, D2, D2P);
        k_prep_b<<<cdiv(DP, TB), TB, 0, stream>>>(b2 + (size_t)l * DD, bp2 + (size_t)l * DP, DD, DP);
    }

    for (int gi = 0; gi < 2; ++gi) {
        const int* x     = (const int*)d_in[gi == 0 ? 0 : 4];
        const int* ei    = (const int*)d_in[gi == 0 ? 1 : 5];
        const int* ea    = (const int*)d_in[gi == 0 ? 2 : 6];
        const int* batch = (const int*)d_in[gi == 0 ? 3 : 7];

        const size_t half = (size_t)GG * TT + (size_t)GG * DD + (size_t)NN * DD;
        float* out_pred = (float*)d_out + (gi == 0 ? 0 : half);
        float* out_grep = out_pred + (size_t)GG * TT;
        float* out_node = out_grep + (size_t)GG * DD;

        k_embed<<<cdiv((long long)MP * (DP / 4), TB), TB, 0, stream>>>(x, xemb1, xemb2, h);

        for (int l = 0; l < LL; ++l) {
            const float* e1l = eemb1 + (size_t)l * 6 * DD;
            const float* e2l = eemb2 + (size_t)l * 3 * DD;

            k_agg_init<<<cdiv((long long)MP * (DP / 4), TB), TB, 0, stream>>>(h, e1l, e2l, agg);
            k_scatter<<<cdiv((long long)EE * (DD / 4), TB), TB, 0, stream>>>(
                ei, ea, h, e1l, e2l, agg);

            // GEMM1: relu(agg[MPxDP] @ W1 + b1) -> h2a[MPxD2P]
            dim3 g1(MP / BM, D2P / BN);
            k_gemm<<<g1, TB, 0, stream>>>(agg, w1t + (size_t)l * D2P * DP,
                                          bp1 + (size_t)l * D2P, h2a, DP, D2P, 1);
            // GEMM2: h2a[MPxD2P] @ W2 + b2 -> agg[MPxDP]
            dim3 g2(MP / BM, DP / BN);
            k_gemm<<<g2, TB, 0, stream>>>(h2a, w2t + (size_t)l * DP * D2P,
                                          bp2 + (size_t)l * DP, agg, D2P, DP, 0);

            k_zero<<<cdiv(1024, TB), TB, 0, stream>>>(gacc, 1024);
            k_bn_partial<<<MP / 64, DP, 0, stream>>>(agg, gacc);
            k_bn_final<<<1, DP, 0, stream>>>(gacc, mu, rstd);

            float* dst = (l == LL - 1) ? out_node : h;
            int ostride = (l == LL - 1) ? DD : DP;
            k_bn_apply<<<cdiv((long long)NN * (DD / 4), TB), TB, 0, stream>>>(
                agg, mu, rstd, gam + (size_t)l * DD, bet + (size_t)l * DD,
                dst, ostride, (l < LL - 1) ? 1 : 0);
        }

        // mean pooling + prediction head
        k_zero<<<cdiv(GG * DD + GG, TB), TB, 0, stream>>>(gsum, GG * DD + GG);
        k_count<<<cdiv(NN, TB), TB, 0, stream>>>(batch, gcnt);
        k_pool_sum<<<cdiv((long long)NN * (DD / 4), TB), TB, 0, stream>>>(
            batch, out_node, gsum);
        k_pool_div<<<cdiv(GG * DD, TB), TB, 0, stream>>>(gsum, gcnt, out_grep);
        k_pred<<<cdiv(GG * TT, TB), TB, 0, stream>>>(out_grep, predW, predb, out_pred);
    }
}
